// CosineContrastiveLoss_72799695667655
// MI455X (gfx1250) — matrix operand order repaired and checked by reading, not compiled, not run on hardware
//
#include <hip/hip_runtime.h>

#define NROWS 8192
#define DDIM  512
#define NTILES (NROWS / 16)   // 512
#define ITPB   4              // i-tiles per block
#define JCHUNK 4              // j-chunks per i-tile (waves per i-tile)
#define JSPAN  (NTILES / JCHUNK)  // 128 j-tiles per wave

typedef __attribute__((ext_vector_type(16))) _Float16 v16h;
typedef __attribute__((ext_vector_type(8)))  float    v8f;
typedef __attribute__((ext_vector_type(4)))  float    f4;
typedef __attribute__((ext_vector_type(4)))  unsigned int v4u;
typedef __attribute__((ext_vector_type(8)))  int      v8i;
typedef __attribute__((ext_vector_type(4)))  int      v4i;

__device__ __constant__ const float kInvTemp = 1.0f / 0.07f;
#define LOG2E 1.4426950408889634f
#define LN2   0.6931471805599453f
#define LDSP  520   // LDS pitch in halfs: 1024B row + 16B TDM pad = 1040B

#if __has_builtin(__builtin_amdgcn_tensor_load_to_lds)
#define USE_TDM 1
#else
#define USE_TDM 0
#endif

#if __has_builtin(__builtin_amdgcn_sched_group_barrier)
#define HAVE_SGB 1
#else
#define HAVE_SGB 0
#endif

// ---------------------------------------------------------------------------
// Kernel 1: row L2-normalize f32[N][D] -> f16[N][D]; one wave32 per row.
// ---------------------------------------------------------------------------
__global__ __launch_bounds__(256) void k_normalize(const float* __restrict__ X,
                                                   _Float16* __restrict__ Y) {
  const int gwave = (blockIdx.x * blockDim.x + threadIdx.x) >> 5;
  const int lane  = threadIdx.x & 31;
  if (gwave >= NROWS) return;

  const f4* src = reinterpret_cast<const f4*>(X + (size_t)gwave * DDIM + lane * 16);
  f4 v0 = src[0], v1 = src[1], v2 = src[2], v3 = src[3];  // 16 floats / lane

  float ss = 0.f;
#pragma unroll
  for (int i = 0; i < 4; ++i) ss += v0[i]*v0[i] + v1[i]*v1[i] + v2[i]*v2[i] + v3[i]*v3[i];
#pragma unroll
  for (int m = 16; m >= 1; m >>= 1) ss += __shfl_xor(ss, m, 32);

  const float inv = 1.0f / fmaxf(sqrtf(ss), 1e-8f);

  union { _Float16 h[16]; f4 f[2]; } o;
#pragma unroll
  for (int i = 0; i < 4; ++i) {
    o.h[i]      = (_Float16)(v0[i] * inv);
    o.h[4 + i]  = (_Float16)(v1[i] * inv);
    o.h[8 + i]  = (_Float16)(v2[i] * inv);
    o.h[12 + i] = (_Float16)(v3[i] * inv);
  }
  f4* dst = reinterpret_cast<f4*>(Y + (size_t)gwave * DDIM + lane * 16);
  dst[0] = o.f[0];
  dst[1] = o.f[1];
}

// ---------------------------------------------------------------------------
// TDM: DMA one 16-row x 512-half tile of T into LDS (with 16B row padding).
// D# per CDNA5 ISA. 2D tensor, data_size=2B.
// 6-arg builtin form: (g0, g1, g2, g3, g4, cpol); groups past 1 unused (2D).
// ---------------------------------------------------------------------------
#if USE_TDM
__device__ __forceinline__ void tdm_load_tile(const _Float16* gsrc,
                                              unsigned lds_byte_addr) {
  const unsigned long long ga = (unsigned long long)(size_t)gsrc;
  v4u g0;
  g0[0] = 1u;                                          // count=1, user mode
  g0[1] = lds_byte_addr;                               // lds_addr[31:0]
  g0[2] = (unsigned)(ga & 0xFFFFFFFFu);                // global_addr[31:0]
  g0[3] = (unsigned)((ga >> 32) & 0x01FFFFFFu)         // global_addr[56:32]
        | (2u << 30);                                  // type=2 ("image")
  v8i g1;
  g1[0] = (int)((1u << 16)      // data_size = 1 -> 2 bytes
              | (1u << 20)      // pad_enable
              | (7u << 22)      // pad_interval: 256 dwords (=1024B row)
              | (3u << 25));    // pad_amount: 4 dwords (=16B)
  g1[1] = (int)(512u << 16);    // tensor_dim0 = 512 (bits 79:48, low half)
  g1[2] = (int)(8192u << 16);   // dim0 hi=0 | tensor_dim1 = 8192 (low half)
  g1[3] = (int)(512u << 16);    // dim1 hi=0 | tile_dim0 = 512
  g1[4] = 16;                   // tile_dim1 = 16, tile_dim2 = 0
  g1[5] = 512;                  // tensor_dim0_stride[31:0] = 512 elements
  g1[6] = 0;                    // stride0 hi | stride1 lo
  g1[7] = 0;
  const v4i g2 = {0, 0, 0, 0};  // 2D: groups 2/3 unused
  const v4i g3 = {0, 0, 0, 0};
  const v8i g4 = {0, 0, 0, 0, 0, 0, 0, 0};  // unused trailing group (6-arg form)
  __builtin_amdgcn_tensor_load_to_lds(g0, g1, g2, g3, g4, 0);
}
#endif

// ---------------------------------------------------------------------------
// Kernel 2: fused WMMA GEMM + streaming logsumexp.
// Block = 512 threads (16 waves) covers 4 i-tiles (64 rows).
//   wave w: i-tile q = w & 3, j-chunk = w >> 2 (128 j-tiles, lockstep).
// TDM path: chunk-leader wave DMAs the next 16KB T-tile to LDS (TENSORcnt)
// while all 4 waves of the chunk consume the current tile via ds_load_b128.
// Each block pulls Tn from L2 once (8 MB); 128 blocks => ~1 GB L2 total.
// sched_group_barrier pins a DS-read/WMMA software pipeline (loads issued
// two K-chunks ahead) so LDS latency overlaps WMMA execution.
// ---------------------------------------------------------------------------
__global__ __launch_bounds__(512) void k_fused(const _Float16* __restrict__ P,
                                               const _Float16* __restrict__ T,
                                               float* __restrict__ rowloss) {
#if USE_TDM
  __shared__ _Float16 bTile[JCHUNK][2][16 * LDSP];   // 4 x 2 x 16.6KB
#endif
  __shared__ float stM[ITPB][16][JCHUNK];
  __shared__ float stS[ITPB][16][JCHUNK];
  __shared__ float diag[ITPB][16];

  const int tid   = threadIdx.x;
  const int wave  = tid >> 5;
  const int lane  = tid & 31;
  const int half  = lane >> 4;
  const int l16   = lane & 15;
  const int q     = wave & (ITPB - 1);     // i-tile slot in block
  const int chunk = wave >> 2;             // j-chunk (0..3)
  const int it    = blockIdx.x * ITPB + q; // global i-tile

  // A fragments (16 K-chunks of 32) -> registers, ISA 16-bit A layout:
  // lane(m=l16, half): K = kc*32 + half*8 + {0..7} and +16.
  union AV { v16h h; f4 f[2]; };
  AV a[16];
  {
    const _Float16* aptr = P + (size_t)(it * 16 + l16) * DDIM + half * 8;
#pragma unroll
    for (int kc = 0; kc < 16; ++kc) {
      a[kc].f[0] = *reinterpret_cast<const f4*>(aptr + kc * 32);
      a[kc].f[1] = *reinterpret_cast<const f4*>(aptr + kc * 32 + 16);
    }
  }

  const float scl = kInvTemp * LOG2E;      // dot -> logit in log2 units
  float m2[8], s2[8];
#pragma unroll
  for (int r = 0; r < 8; ++r) { m2[r] = -1e30f; s2[r] = 0.f; }

  const int jlo = chunk * JSPAN;

#if USE_TDM
  // Prime buffer 0 with the chunk's first tile (leader wave only).
  if (q == 0) {
    tdm_load_tile(T + (size_t)jlo * 16 * DDIM,
                  (unsigned)(size_t)&bTile[chunk][0][0]);  // low32 = LDS addr
    __builtin_amdgcn_s_wait_tensorcnt(0);
  }
  __syncthreads();
#endif

  for (int s = 0; s < JSPAN; ++s) {
    const int jt = jlo + s;
#if USE_TDM
    const int buf = s & 1;
    if (q == 0 && s + 1 < JSPAN)   // async-DMA next tile into other buffer
      tdm_load_tile(T + (size_t)(jt + 1) * 16 * DDIM,
                    (unsigned)(size_t)&bTile[chunk][buf ^ 1][0]);
    const _Float16* brow = &bTile[chunk][buf][l16 * LDSP + half * 8];
#else
    if (s + 1 < JSPAN)
      __builtin_prefetch(T + (size_t)((jt + 1) * 16 + l16) * DDIM, 0, 1);
    const _Float16* brow = T + (size_t)(jt * 16 + l16) * DDIM + half * 8;
#endif

    // K = 512 in chunks of 32; B double-buffered in registers.
    v8f c = {};
    union BV { v16h h; f4 f[2]; } bb[2];
    bb[0].f[0] = *reinterpret_cast<const f4*>(brow);
    bb[0].f[1] = *reinterpret_cast<const f4*>(brow + 16);
#pragma unroll
    for (int kc = 0; kc < 16; ++kc) {
      if (kc + 1 < 16) {
        bb[(kc + 1) & 1].f[0] = *reinterpret_cast<const f4*>(brow + (kc + 1) * 32);
        bb[(kc + 1) & 1].f[1] = *reinterpret_cast<const f4*>(brow + (kc + 1) * 32 + 16);
      }
      c = __builtin_amdgcn_wmma_f32_16x16x32_f16(false, a[kc].h, false,
                                                 bb[kc & 1].h,
                                                 (short)0, c, false, false);
    }

#if HAVE_SGB && USE_TDM
    // Enforce the pipeline shape: 4 DS reads up front (2 chunks in flight),
    // then [1 WMMA, 2 DS reads] x14, then the last 2 WMMAs. Loads land two
    // chunks ahead of their consuming WMMA -> s_wait_dscnt > 0.
    __builtin_amdgcn_sched_group_barrier(0x200, 4, 0);   // DS read x4
#pragma unroll
    for (int g = 0; g < 14; ++g) {
      __builtin_amdgcn_sched_group_barrier(0x008, 1, 0); // WMMA x1
      __builtin_amdgcn_sched_group_barrier(0x200, 2, 0); // DS read x2
    }
    __builtin_amdgcn_sched_group_barrier(0x008, 2, 0);   // WMMA x2
#endif

    // Online logsumexp (log2 domain): lane owns rows m = r + 8*half,
    // column j = jt*16 + l16.
#pragma unroll
    for (int r = 0; r < 8; ++r) {
      const float v = c[r] * scl;
      const float m = fmaxf(m2[r], v);
      s2[r] = s2[r] * __builtin_exp2f(m2[r] - m) + __builtin_exp2f(v - m);
      m2[r] = m;
    }

    // Capture diagonal dot when this wave owns the i==j tile.
    if (jt == it) {
#pragma unroll
      for (int r = 0; r < 8; ++r)
        if (l16 == r + 8 * half) diag[q][l16] = c[r];   // raw dot, row m=l16
    }

#if USE_TDM
    if (q == 0 && s + 1 < JSPAN) __builtin_amdgcn_s_wait_tensorcnt(0);
    __syncthreads();   // publish next buffer; all chunk waves advance together
#endif
  }

  // Merge softmax state across the 16 lanes of each half (rows differ by half).
#pragma unroll
  for (int r = 0; r < 8; ++r) {
#pragma unroll
    for (int d = 1; d < 16; d <<= 1) {
      const float om = __shfl_xor(m2[r], d, 32);
      const float os = __shfl_xor(s2[r], d, 32);
      const float m  = fmaxf(m2[r], om);
      s2[r] = s2[r] * __builtin_exp2f(m2[r] - m) + os * __builtin_exp2f(om - m);
      m2[r] = m;
    }
    if (l16 == 0) {
      stM[q][r + 8 * half][chunk] = m2[r];
      stS[q][r + 8 * half][chunk] = s2[r];
    }
  }
  __syncthreads();

  // One wave per i-tile (chunk-0 waves) merges the 4 j-chunk states per row.
  if (chunk == 0 && lane < 16) {
    float M = stM[q][lane][0], S = stS[q][lane][0];
#pragma unroll
    for (int w = 1; w < JCHUNK; ++w) {
      const float om = stM[q][lane][w], os = stS[q][lane][w];
      const float m  = fmaxf(M, om);
      S = S * __builtin_exp2f(M - m) + os * __builtin_exp2f(om - m);
      M = m;
    }
    const float lse = (M + __builtin_log2f(S)) * LN2;
    const float pos = diag[q][lane] * kInvTemp;
    rowloss[it * 16 + lane] = lse - pos;
  }
}

// ---------------------------------------------------------------------------
// Kernel 3: deterministic fixed-order mean of the 8192 row losses.
// ---------------------------------------------------------------------------
__global__ __launch_bounds__(256) void k_reduce(const float* __restrict__ rl,
                                                float* __restrict__ out) {
  __shared__ float sh[256];
  float s = 0.f;
  for (int i = threadIdx.x; i < NROWS; i += 256) s += rl[i];
  sh[threadIdx.x] = s;
  __syncthreads();
#pragma unroll
  for (int k = 128; k >= 1; k >>= 1) {
    if (threadIdx.x < k) sh[threadIdx.x] += sh[threadIdx.x + k];
    __syncthreads();
  }
  if (threadIdx.x == 0) out[0] = sh[0] * (1.0f / (float)NROWS);
}

// ---------------------------------------------------------------------------
extern "C" void kernel_launch(void* const* d_in, const int* in_sizes, int n_in,
                              void* d_out, int out_size, void* d_ws, size_t ws_size,
                              hipStream_t stream) {
  const float* P32 = (const float*)d_in[0];   // projected_proteins [8192,512] f32
  const float* T32 = (const float*)d_in[1];   // projected_tokens   [8192,512] f32

  _Float16* Pn = (_Float16*)d_ws;                          // 8 MB
  _Float16* Tn = Pn + (size_t)NROWS * DDIM;                // 8 MB
  float* rowloss = (float*)(Tn + (size_t)NROWS * DDIM);    // 32 KB

  k_normalize<<<NROWS / 8, 256, 0, stream>>>(P32, Pn);
  k_normalize<<<NROWS / 8, 256, 0, stream>>>(T32, Tn);
  k_fused<<<NTILES / ITPB, 512, 0, stream>>>(Pn, Tn, rowloss);
  k_reduce<<<1, 256, 0, stream>>>(rowloss, (float*)d_out);
}
